// MultiheadSelfAttention_81655918232006
// MI455X (gfx1250) — matrix operand order repaired and checked by reading, not compiled, not run on hardware
//
#include <hip/hip_runtime.h>

typedef __attribute__((ext_vector_type(16))) __bf16 v16bf;
typedef __attribute__((ext_vector_type(8)))  __bf16 v8bf;
typedef __attribute__((ext_vector_type(8)))  float  v8f;

#define DM    1024
#define HEADS 16
#define DK    64
#define SEQ   2048
#define BATCH 2
#define MTOT  (BATCH*SEQ)

// ---------- WMMA helpers (CDNA5 16x16x32 bf16, f32 accum) ----------

__device__ __forceinline__ v8f wmma_bf16(v16bf a, v16bf b, v8f c) {
  return __builtin_amdgcn_wmma_f32_16x16x32_bf16(false, a, false, b, (short)0, c,
                                                 false, false);
}

// A matrix 16x32 (MxK), row-major source with leading dim ld.
// Lanes 0-15: M=lane, K = {kb..kb+7, kb+16..kb+23}, kb=0; lanes 16-31: kb=8.
__device__ __forceinline__ v16bf load_a_frag(const __bf16* __restrict__ src,
                                             int row0, int k0, int ld, int lane) {
  int m  = lane & 15;
  int kb = (lane >> 4) * 8;
  const __bf16* p = src + (size_t)(row0 + m) * ld + k0 + kb;
  union { v16bf v; v8bf h[2]; } u;
  u.h[0] = *(const v8bf*)(p);
  u.h[1] = *(const v8bf*)(p + 16);
  return u.v;
}

// A fragment from LDS (row-major 16x32 tile, stride 32)
__device__ __forceinline__ v16bf load_a_lds(const __bf16* p0, int lane) {
  int m  = lane & 15;
  int kb = (lane >> 4) * 8;
  const __bf16* p = p0 + m * 32 + kb;
  union { v16bf v; v8bf h[2]; } u;
  u.h[0] = *(const v8bf*)(p);
  u.h[1] = *(const v8bf*)(p + 16);
  return u.v;
}

// B matrix 32x16 (KxN), where B[k][n] = src[n][k] (src row-major, leading dim ld).
__device__ __forceinline__ v16bf load_b_frag(const __bf16* __restrict__ src,
                                             int n0, int k0, int ld, int lane) {
  int n    = lane & 15;
  int koff = (lane >> 4) * 8;
  const __bf16* p = src + (size_t)(n0 + n) * ld + k0 + koff;
  union { v16bf v; v8bf h[2]; } u;
  u.h[0] = *(const v8bf*)(p);
  u.h[1] = *(const v8bf*)(p + 16);
  return u.v;
}

// ---------- elementwise prep kernels ----------

__global__ void f32_to_bf16(const float* __restrict__ src, __bf16* __restrict__ dst,
                            size_t n) {
  size_t i = (size_t)blockIdx.x * blockDim.x + threadIdx.x;
  if (i < n) dst[i] = (__bf16)src[i];
}

// RoPE on fp32 Q/K (merged layout [b*s][h*64+d]) -> bf16. One thread per pair.
__global__ void rope_to_bf16(const float* __restrict__ Qf, const float* __restrict__ Kf,
                             const int* __restrict__ pos,
                             __bf16* __restrict__ Qb, __bf16* __restrict__ Kb) {
  size_t idx = (size_t)blockIdx.x * blockDim.x + threadIdx.x;  // MTOT*512 pairs
  if (idx >= (size_t)MTOT * 512) return;
  int bs = (int)(idx >> 9);
  int p  = (int)(idx & 511);
  int s  = bs & (SEQ - 1);
  int i  = p & 31;  // pair index within head (0..31)
  float ang = (float)pos[s] * __powf(10000.0f, -(float)i * (1.0f / 32.0f));
  float sn, cs;
  __sincosf(ang, &sn, &cs);
  size_t e = (size_t)bs * DM + 2 * p;
  float q1 = Qf[e], q2 = Qf[e + 1];
  Qb[e]     = (__bf16)(q1 * cs - q2 * sn);
  Qb[e + 1] = (__bf16)(q1 * sn + q2 * cs);
  float k1 = Kf[e], k2 = Kf[e + 1];
  Kb[e]     = (__bf16)(k1 * cs - k2 * sn);
  Kb[e + 1] = (__bf16)(k1 * sn + k2 * cs);
}

// V fp32 [b*s][h*64+d] -> bf16 Vt [b][h][d][s]
__global__ void v_transpose_bf16(const float* __restrict__ Vf, __bf16* __restrict__ Vt) {
  size_t i = (size_t)blockIdx.x * blockDim.x + threadIdx.x;
  if (i >= (size_t)MTOT * DM) return;
  int m = (int)(i / DM);   // b*SEQ + s
  int c = (int)(i % DM);   // h*64 + d
  int b = m / SEQ, s = m % SEQ;
  Vt[(((size_t)b * HEADS + (c >> 6)) * DK + (c & 63)) * SEQ + s] = (__bf16)Vf[i];
}

// ---------- bf16 GEMM: C[m][n] = sum_k A[m][k] * B[n][k], fp32 out ----------
// One wave per 32x64 tile: 8 WMMAs per 6 fragment loads each K-step.

__global__ void __launch_bounds__(32)
gemm_bf16_nt(const __bf16* __restrict__ A, const __bf16* __restrict__ B,
             float* __restrict__ C, int M, int N, int K) {
  int lane = threadIdx.x;
  int n0 = blockIdx.x * 64;
  int m0 = blockIdx.y * 32;
  v8f c[2][4] = {};
  for (int k0 = 0; k0 < K; k0 += 32) {
    v16bf a0 = load_a_frag(A, m0,      k0, K, lane);
    v16bf a1 = load_a_frag(A, m0 + 16, k0, K, lane);
#pragma unroll
    for (int j = 0; j < 4; ++j) {
      v16bf b = load_b_frag(B, n0 + 16 * j, k0, K, lane);
      c[0][j] = wmma_bf16(a0, b, c[0][j]);
      c[1][j] = wmma_bf16(a1, b, c[1][j]);
    }
  }
  int col  = lane & 15;
  int half = lane >> 4;
#pragma unroll
  for (int j = 0; j < 4; ++j) {
#pragma unroll
    for (int r = 0; r < 8; ++r) {
      int row = r + 8 * half;
      C[(size_t)(m0 + row) * N + n0 + 16 * j + col]      = c[0][j][r];
      C[(size_t)(m0 + 16 + row) * N + n0 + 16 * j + col] = c[1][j][r];
    }
  }
}

// ---------- Flash attention: one wave handles 32 query rows of one (b,h) ----------
// blockIdx.x is reversed so the heavy (large-q0, long-loop) blocks launch first.

__global__ void __launch_bounds__(32)
attn_flash(const __bf16* __restrict__ Qb, const __bf16* __restrict__ Kb,
           const __bf16* __restrict__ Vt, __bf16* __restrict__ Ob) {
  int lane = threadIdx.x;
  int q0 = SEQ - 32 * (blockIdx.x + 1);  // heavy blocks first
  int h  = blockIdx.y;
  int b  = blockIdx.z;

  const __bf16* Qh = Qb + (size_t)b * SEQ * DM + h * DK;       // ld = DM
  const __bf16* Kh = Kb + (size_t)b * SEQ * DM + h * DK;       // ld = DM
  const __bf16* Vh = Vt + ((size_t)b * HEADS + h) * DK * SEQ;  // rows = d, ld = SEQ

  // Q fragments: [qtile][kfrag]; qtile t covers rows q0+16t, kfrag j covers d 32j..32j+31
  v16bf qa[2][2];
#pragma unroll
  for (int t = 0; t < 2; ++t) {
    qa[t][0] = load_a_frag(Qh, q0 + 16 * t, 0,  DM, lane);
    qa[t][1] = load_a_frag(Qh, q0 + 16 * t, 32, DM, lane);
  }

  v8f o[2][4] = {};          // [qtile][dtile]
  float rm[2][8], rs[2][8];  // running max / sum per row
#pragma unroll
  for (int t = 0; t < 2; ++t)
#pragma unroll
    for (int r = 0; r < 8; ++r) { rm[t][r] = -3.0e38f; rs[t][r] = 0.0f; }

  __shared__ __bf16 pl[2 * 16 * 32];  // two 16x32 P tiles

  int col  = lane & 15;
  int half = lane >> 4;
  const float scale = 0.125f;  // 1/sqrt(64)

  for (int kv = 0; kv < q0 + 32; kv += 32) {
    // ---- scores: sc[qtile][ktile], K fragments reused across both q tiles ----
    v8f sc[2][2] = {};
    {
      v16bf kb;
      kb = load_b_frag(Kh, kv, 0, DM, lane);
      sc[0][0] = wmma_bf16(qa[0][0], kb, sc[0][0]);
      sc[1][0] = wmma_bf16(qa[1][0], kb, sc[1][0]);
      kb = load_b_frag(Kh, kv, 32, DM, lane);
      sc[0][0] = wmma_bf16(qa[0][1], kb, sc[0][0]);
      sc[1][0] = wmma_bf16(qa[1][1], kb, sc[1][0]);
      kb = load_b_frag(Kh, kv + 16, 0, DM, lane);
      sc[0][1] = wmma_bf16(qa[0][0], kb, sc[0][1]);
      sc[1][1] = wmma_bf16(qa[1][0], kb, sc[1][1]);
      kb = load_b_frag(Kh, kv + 16, 32, DM, lane);
      sc[0][1] = wmma_bf16(qa[0][1], kb, sc[0][1]);
      sc[1][1] = wmma_bf16(qa[1][1], kb, sc[1][1]);
    }

    // ---- online softmax per q tile ----
#pragma unroll
    for (int t = 0; t < 2; ++t) {
      float p0[8], p1[8];
#pragma unroll
      for (int r = 0; r < 8; ++r) {
        int row = q0 + 16 * t + r + 8 * half;  // global query row
        int c0  = kv + col;
        int c1  = c0 + 16;
        float v0 = (c0 <= row) ? sc[t][0][r] * scale : -1.0e9f;
        float v1 = (c1 <= row) ? sc[t][1][r] * scale : -1.0e9f;
        float tm = fmaxf(v0, v1);
        tm = fmaxf(tm, __shfl_xor(tm, 1));
        tm = fmaxf(tm, __shfl_xor(tm, 2));
        tm = fmaxf(tm, __shfl_xor(tm, 4));
        tm = fmaxf(tm, __shfl_xor(tm, 8));
        float mn = fmaxf(rm[t][r], tm);
        float a  = __expf(rm[t][r] - mn);
        float e0 = __expf(v0 - mn);
        float e1 = __expf(v1 - mn);
        float ts = e0 + e1;
        ts += __shfl_xor(ts, 1);
        ts += __shfl_xor(ts, 2);
        ts += __shfl_xor(ts, 4);
        ts += __shfl_xor(ts, 8);
        rs[t][r] = rs[t][r] * a + ts;
        rm[t][r] = mn;
        p0[r] = e0;
        p1[r] = e1;
        o[t][0][r] *= a; o[t][1][r] *= a; o[t][2][r] *= a; o[t][3][r] *= a;
      }
      // C/D-layout P -> row-major LDS tile
#pragma unroll
      for (int r = 0; r < 8; ++r) {
        int row = r + 8 * half;
        pl[t * 512 + row * 32 + col]      = (__bf16)p0[r];
        pl[t * 512 + row * 32 + 16 + col] = (__bf16)p1[r];
      }
    }
    __syncthreads();
    v16bf pa0 = load_a_lds(&pl[0],   lane);
    v16bf pa1 = load_a_lds(&pl[512], lane);
    __syncthreads();

    // ---- P @ V : V fragments reused across both q tiles ----
#pragma unroll
    for (int j = 0; j < 4; ++j) {
      v16bf vb = load_b_frag(Vh, 16 * j, kv, SEQ, lane);
      o[0][j] = wmma_bf16(pa0, vb, o[0][j]);
      o[1][j] = wmma_bf16(pa1, vb, o[1][j]);
    }
  }

#pragma unroll
  for (int t = 0; t < 2; ++t) {
#pragma unroll
    for (int r = 0; r < 8; ++r) {
      float inv = 1.0f / rs[t][r];
      size_t base = (size_t)(b * SEQ + q0 + 16 * t + r + 8 * half) * DM + h * DK;
      Ob[base + col]      = (__bf16)(o[t][0][r] * inv);
      Ob[base + 16 + col] = (__bf16)(o[t][1][r] * inv);
      Ob[base + 32 + col] = (__bf16)(o[t][2][r] * inv);
      Ob[base + 48 + col] = (__bf16)(o[t][3][r] * inv);
    }
  }
}

// ---------- host ----------

extern "C" void kernel_launch(void* const* d_in, const int* in_sizes, int n_in,
                              void* d_out, int out_size, void* d_ws, size_t ws_size,
                              hipStream_t stream) {
  (void)in_sizes; (void)n_in; (void)out_size; (void)ws_size;
  const float* X  = (const float*)d_in[0];
  const int*  pos = (const int*)d_in[1];
  const float* Wq = (const float*)d_in[2];
  const float* Wk = (const float*)d_in[3];
  const float* Wv = (const float*)d_in[4];
  const float* Wo = (const float*)d_in[5];
  float* out = (float*)d_out;

  char* ws = (char*)d_ws;
  size_t off = 0;
  auto alloc = [&](size_t bytes) -> char* {
    char* p = ws + off;
    off += (bytes + 255) & ~(size_t)255;
    return p;
  };

  __bf16* Xb  = (__bf16*)alloc((size_t)MTOT * DM * 2);
  __bf16* Wqb = (__bf16*)alloc((size_t)DM * DM * 2);
  __bf16* Wkb = (__bf16*)alloc((size_t)DM * DM * 2);
  __bf16* Wvb = (__bf16*)alloc((size_t)DM * DM * 2);
  __bf16* Wob = (__bf16*)alloc((size_t)DM * DM * 2);
  float*  Qf  = (float*)alloc((size_t)MTOT * DM * 4);
  float*  Kf  = (float*)alloc((size_t)MTOT * DM * 4);
  float*  Vf  = (float*)alloc((size_t)MTOT * DM * 4);
  __bf16* Qbr = (__bf16*)alloc((size_t)MTOT * DM * 2);
  __bf16* Kbr = (__bf16*)alloc((size_t)MTOT * DM * 2);
  __bf16* Vtb = (__bf16*)alloc((size_t)MTOT * DM * 2);
  __bf16* Obf = (__bf16*)alloc((size_t)MTOT * DM * 2);

  size_t nX = (size_t)MTOT * DM;
  size_t nW = (size_t)DM * DM;
  f32_to_bf16<<<(unsigned)((nX + 255) / 256), 256, 0, stream>>>(X,  Xb,  nX);
  f32_to_bf16<<<(unsigned)((nW + 255) / 256), 256, 0, stream>>>(Wq, Wqb, nW);
  f32_to_bf16<<<(unsigned)((nW + 255) / 256), 256, 0, stream>>>(Wk, Wkb, nW);
  f32_to_bf16<<<(unsigned)((nW + 255) / 256), 256, 0, stream>>>(Wv, Wvb, nW);
  f32_to_bf16<<<(unsigned)((nW + 255) / 256), 256, 0, stream>>>(Wo, Wob, nW);

  dim3 gg(DM / 64, MTOT / 32);
  gemm_bf16_nt<<<gg, 32, 0, stream>>>(Xb, Wqb, Qf, MTOT, DM, DM);
  gemm_bf16_nt<<<gg, 32, 0, stream>>>(Xb, Wkb, Kf, MTOT, DM, DM);
  gemm_bf16_nt<<<gg, 32, 0, stream>>>(Xb, Wvb, Vf, MTOT, DM, DM);

  size_t nR = (size_t)MTOT * 512;
  rope_to_bf16<<<(unsigned)((nR + 255) / 256), 256, 0, stream>>>(Qf, Kf, pos, Qbr, Kbr);
  v_transpose_bf16<<<(unsigned)((nX + 255) / 256), 256, 0, stream>>>(Vf, Vtb);

  attn_flash<<<dim3(SEQ / 32, HEADS, BATCH), 32, 0, stream>>>(Qbr, Kbr, Vtb, Obf);

  gemm_bf16_nt<<<gg, 32, 0, stream>>>(Obf, Wob, out, MTOT, DM, DM);
}
